// KeyedConv2d_21895743275543
// MI455X (gfx1250) — matrix operand order repaired and checked by reading, not compiled, not run on hardware
//
#include <hip/hip_runtime.h>

// ---------------------------------------------------------------------------
// KeyedConv2d: out = A @ T @ Ainv @ x
//   GEMM1 (8193x8193 @ 8193x1024) -> conv-apply -> GEMM2 (4097x4097 @ 4097x1024)
// GEMMs: V_WMMA_F32_16X16X32_BF16, f32 inputs split in-register to bf16 hi/lo,
// C ~= Ah*Bh + Ah*Bl + Al*Bh (f32 accumulate). Block 128x128x32, 16 waves,
// wave tile 32x32. Edge handling hoisted to separate kernel launches.
// ---------------------------------------------------------------------------

typedef __attribute__((ext_vector_type(16))) __bf16 v16bf;
typedef __attribute__((ext_vector_type(8)))  __bf16 v8bf;
typedef __attribute__((ext_vector_type(4)))  __bf16 v4bf;
typedef __attribute__((ext_vector_type(8)))  float  v8f;
typedef __attribute__((ext_vector_type(4)))  float  v4f;

#define GEMM_N       1024      // fixed ldb/ldc
#define BM           128
#define BN           128
#define BK           32
#define LDA_PAD      40        // bf16/row: 80B rows (16B-aligned, bank-spread)
#define LDB_PAD      136       // bf16/row: 272B rows (16B-aligned, bank-spread)
#define GEMM_THREADS 512       // 16 waves: 4(M) x 4(N), 32x32 per wave
#define AQ           ((BM * BK) / (4 * GEMM_THREADS))   // A quads per thread = 2
#define BQ           ((BK * BN) / (4 * GEMM_THREADS))   // B quads per thread = 2

static __device__ __forceinline__ v16bf cat8(v8bf a, v8bf b) {
    return __builtin_shufflevector(a, b, 0,1,2,3,4,5,6,7,8,9,10,11,12,13,14,15);
}

// Stage one BMxBK A tile and BKxBN B tile into LDS as bf16 hi/lo.
// All global loads are issued first (kept in flight together), then converted.
// GM: guard A rows against M.  GK: guard against K (tail step).
template<bool GM, bool GK>
static __device__ __forceinline__ void stage_tiles(
    const float* __restrict__ Ag, const float* __restrict__ Bg,
    __bf16* lAh, __bf16* lAl, __bf16* lBh, __bf16* lBl,
    int tid, int m0, int n0, int kb, int M, int K) {
    float fa[AQ][4];
    v4f   fb[BQ];
    int   ar[AQ], ak[AQ], br[BQ], bn[BQ];

    // ---- issue all A loads (4 consecutive k, 4B-aligned; vectorizer fuses) ----
    #pragma unroll
    for (int q = 0; q < AQ; ++q) {
        int quad = q * GEMM_THREADS + tid;
        ar[q] = quad >> 3;                 // 8 quads per 32-wide row
        ak[q] = (quad & 7) << 2;
        int gm = m0 + ar[q];
        const float* src = Ag + (size_t)gm * (size_t)K + (size_t)(kb + ak[q]);
        #pragma unroll
        for (int j = 0; j < 4; ++j) {
            bool ok = true;
            if (GM) ok = ok && (gm < M);
            if (GK) ok = ok && ((kb + ak[q] + j) < K);
            fa[q][j] = ok ? src[j] : 0.f;
        }
    }
    // ---- issue all B loads (float4, ldb=1024 -> 16B aligned) ----
    #pragma unroll
    for (int q = 0; q < BQ; ++q) {
        int quad = q * GEMM_THREADS + tid;
        br[q] = quad >> 5;                 // 32 quads per 128-wide row
        bn[q] = (quad & 31) << 2;
        int gk = kb + br[q];
        fb[q] = (v4f){0.f, 0.f, 0.f, 0.f};
        if (!GK || gk < K)
            fb[q] = *(const v4f*)(Bg + (size_t)gk * GEMM_N + (size_t)(n0 + bn[q]));
    }
    // ---- convert + store A ----
    #pragma unroll
    for (int q = 0; q < AQ; ++q) {
        v4bf h, l;
        #pragma unroll
        for (int j = 0; j < 4; ++j) {
            __bf16 hh = (__bf16)fa[q][j];
            h[j] = hh;
            l[j] = (__bf16)(fa[q][j] - (float)hh);
        }
        *(v4bf*)&lAh[ar[q] * LDA_PAD + ak[q]] = h;
        *(v4bf*)&lAl[ar[q] * LDA_PAD + ak[q]] = l;
    }
    // ---- convert + store B ----
    #pragma unroll
    for (int q = 0; q < BQ; ++q) {
        v4bf h, l;
        #pragma unroll
        for (int j = 0; j < 4; ++j) {
            __bf16 hh = (__bf16)fb[q][j];
            h[j] = hh;
            l[j] = (__bf16)(fb[q][j] - (float)hh);
        }
        *(v4bf*)&lBh[br[q] * LDB_PAD + bn[q]] = h;
        *(v4bf*)&lBl[br[q] * LDB_PAD + bn[q]] = l;
    }
}

// C[M x 1024] = A[M x K] (row-major, lda=K) @ B[K x 1024] (row-major).
// EDGE=false: all BM rows of this block are valid (no M guards at all).
// EDGE=true : row guards active (used only for the last partial row-block).
template<bool EDGE>
__global__ __launch_bounds__(GEMM_THREADS)
void keyed_gemm_bf16x3(const float* __restrict__ Ag,
                       const float* __restrict__ Bg,
                       float* __restrict__ Cg,
                       int M, int K, int mBase) {
    __shared__ __align__(16) __bf16 lAh[BM * LDA_PAD];
    __shared__ __align__(16) __bf16 lAl[BM * LDA_PAD];
    __shared__ __align__(16) __bf16 lBh[BK * LDB_PAD];
    __shared__ __align__(16) __bf16 lBl[BK * LDB_PAD];

    const int tid  = threadIdx.x;
    const int lane = tid & 31;
    const int wid  = tid >> 5;
    const int wm   = wid >> 2;          // 0..3 -> 32-row band
    const int wn   = wid & 3;           // 0..3 -> 32-col band
    const int lh   = lane >> 4;
    const int lm   = lane & 15;

    const int m0 = mBase + blockIdx.y * BM;
    const int n0 = blockIdx.x * BN;

    v8f acc[2][2];
    #pragma unroll
    for (int i = 0; i < 2; ++i)
        #pragma unroll
        for (int j = 0; j < 2; ++j)
            acc[i][j] = (v8f){0.f,0.f,0.f,0.f,0.f,0.f,0.f,0.f};

    auto compute_step = [&]() {
        // A 16x32 frag: lane = row (lane&15); K runs [8lh..8lh+7], [16+8lh..23+8lh]
        v16bf ah[2], al[2];
        #pragma unroll
        for (int mt = 0; mt < 2; ++mt) {
            int off = (wm * 32 + mt * 16 + lm) * LDA_PAD + 8 * lh;
            ah[mt] = cat8(*(const v8bf*)&lAh[off], *(const v8bf*)&lAh[off + 16]);
            al[mt] = cat8(*(const v8bf*)&lAl[off], *(const v8bf*)&lAl[off + 16]);
        }
        // B 32x16 frag: lane = col (lane&15); K runs [8lh..8lh+7], [16+8lh..23+8lh]
        v16bf bh[2], bl[2];
        #pragma unroll
        for (int nt = 0; nt < 2; ++nt) {
            int col = wn * 32 + nt * 16 + lm;
            #pragma unroll
            for (int s = 0; s < 16; ++s) {
                int k = (s < 8) ? (8 * lh + s) : (16 + 8 * lh + (s - 8));
                bh[nt][s] = lBh[k * LDB_PAD + col];
                bl[nt][s] = lBl[k * LDB_PAD + col];
            }
        }
        #pragma unroll
        for (int mt = 0; mt < 2; ++mt)
            #pragma unroll
            for (int nt = 0; nt < 2; ++nt) {
                acc[mt][nt] = __builtin_amdgcn_wmma_f32_16x16x32_bf16(
                    false, ah[mt], false, bh[nt], (short)0, acc[mt][nt], false, false);
                acc[mt][nt] = __builtin_amdgcn_wmma_f32_16x16x32_bf16(
                    false, ah[mt], false, bl[nt], (short)0, acc[mt][nt], false, false);
                acc[mt][nt] = __builtin_amdgcn_wmma_f32_16x16x32_bf16(
                    false, al[mt], false, bh[nt], (short)0, acc[mt][nt], false, false);
            }
    };

    const int kfull = K - (K & (BK - 1));
    for (int kb = 0; kb < kfull; kb += BK) {
        stage_tiles<EDGE, false>(Ag, Bg, lAh, lAl, lBh, lBl, tid, m0, n0, kb, M, K);
        if (!EDGE && kb + BK < kfull) {   // prefetch next K-step (speculative)
            __builtin_prefetch(Ag + (size_t)(m0 + (tid >> 3)) * (size_t)K +
                                   (size_t)(kb + BK + ((tid & 7) << 2)), 0, 0);
            __builtin_prefetch(Bg + (size_t)(kb + BK + (tid >> 5)) * GEMM_N +
                                   (size_t)(n0 + ((tid & 31) << 2)), 0, 0);
        }
        __syncthreads();
        compute_step();
        __syncthreads();
    }
    if (kfull < K) {   // K tail (K % 32 == 1 here)
        stage_tiles<EDGE, true>(Ag, Bg, lAh, lAl, lBh, lBl, tid, m0, n0, kfull, M, K);
        __syncthreads();
        compute_step();
    }

    // C/D 16x16 f32 frag: VGPR r -> row r + 8*lh; lane&15 -> col
    #pragma unroll
    for (int mt = 0; mt < 2; ++mt)
        #pragma unroll
        for (int nt = 0; nt < 2; ++nt) {
            int col   = n0 + wn * 32 + nt * 16 + lm;
            int rbase = m0 + wm * 32 + mt * 16 + 8 * lh;
            #pragma unroll
            for (int r = 0; r < 8; ++r) {
                int row = rbase + r;
                if (!EDGE || row < M)
                    Cg[(size_t)row * GEMM_N + (size_t)col] = acc[mt][nt][r];
            }
        }
}

// ---------------------------------------------------------------------------
// Toeplitz-conv apply: t = [[conv,bias],[0,1]] @ z
// z: (8193,1024) row-major; t: (4097,1024) row-major. One block per column n.
// ---------------------------------------------------------------------------
#define CIN   8
#define COUT  16
#define UIN   32
#define UOUT  16

__global__ __launch_bounds__(256)
void keyed_conv_apply(const float* __restrict__ z,
                      const float* __restrict__ w,
                      const float* __restrict__ b,
                      float* __restrict__ t) {
    __shared__ float xcol[CIN * UIN * UIN];   // 8192 f32 = 32KB
    __shared__ float wsm[COUT * CIN * 3 * 3];
    __shared__ float bsm[COUT];

    const int n   = blockIdx.x;
    const int tid = threadIdx.x;

    for (int i = tid; i < CIN * UIN * UIN; i += 256)
        xcol[i] = z[(size_t)i * GEMM_N + n];
    for (int i = tid; i < COUT * CIN * 9; i += 256)
        wsm[i] = w[i];
    if (tid < COUT) bsm[tid] = b[tid];
    const float hom = z[(size_t)(CIN * UIN * UIN) * GEMM_N + n];
    __syncthreads();

    // SAME pad for in=32,k=3,s=2: pad_lo=0, pad_hi=1
    for (int o = tid; o < COUT * UOUT * UOUT; o += 256) {
        int co = o >> 8;
        int uo = (o >> 4) & 15;
        int vo = o & 15;
        float accv = bsm[co] * hom;
        #pragma unroll
        for (int ci = 0; ci < CIN; ++ci)
            #pragma unroll
            for (int ku = 0; ku < 3; ++ku) {
                int u = uo * 2 + ku;
                if (u < UIN) {
                    #pragma unroll
                    for (int kv = 0; kv < 3; ++kv) {
                        int v = vo * 2 + kv;
                        if (v < UIN)
                            accv += xcol[ci * (UIN * UIN) + u * UIN + v] *
                                    wsm[((co * CIN + ci) * 3 + ku) * 3 + kv];
                    }
                }
            }
        t[(size_t)o * GEMM_N + n] = accv;
    }
    if (tid == 0)
        t[(size_t)(COUT * UOUT * UOUT) * GEMM_N + n] = hom;
}

// ---------------------------------------------------------------------------
static void launch_gemm(const float* A, const float* B, float* C,
                        int M, int K, hipStream_t stream) {
    int mFull = M / BM;                       // full 128-row blocks
    if (mFull > 0) {
        dim3 grid(GEMM_N / BN, mFull);
        keyed_gemm_bf16x3<false><<<grid, GEMM_THREADS, 0, stream>>>(A, B, C, M, K, 0);
    }
    if (M % BM) {                             // partial last row-block
        dim3 grid(GEMM_N / BN, 1);
        keyed_gemm_bf16x3<true><<<grid, GEMM_THREADS, 0, stream>>>(A, B, C, M, K, mFull * BM);
    }
}

extern "C" void kernel_launch(void* const* d_in, const int* in_sizes, int n_in,
                              void* d_out, int out_size, void* d_ws, size_t ws_size,
                              hipStream_t stream) {
    (void)in_sizes; (void)n_in; (void)out_size; (void)ws_size;

    const float* w    = (const float*)d_in[0];   // (16,8,3,3)
    const float* b    = (const float*)d_in[1];   // (16,)
    const float* Amat = (const float*)d_in[2];   // (4097,4097)
    const float* Ainv = (const float*)d_in[3];   // (8193,8193)
    const float* x    = (const float*)d_in[4];   // (8193,1024)
    float*       out  = (float*)d_out;           // (4097,1024)

    const int IN_DIM  = 8193;
    const int OUT_DIM = 4097;

    float* z = (float*)d_ws;                     // (8193,1024) = 33.6 MB
    float* t = z + (size_t)IN_DIM * GEMM_N;      // (4097,1024) = 16.8 MB

    launch_gemm(Ainv, x, z, IN_DIM, IN_DIM, stream);          // z = Ainv @ x
    keyed_conv_apply<<<GEMM_N, 256, 0, stream>>>(z, w, b, t); // t = T @ z
    launch_gemm(Amat, t, out, OUT_DIM, OUT_DIM, stream);      // out = A @ t
}
